// RecurrentGCN_69587060130083
// MI455X (gfx1250) — matrix operand order repaired and checked by reading, not compiled, not run on hardware
//
#include <hip/hip_runtime.h>
#include <hip/hip_bf16.h>

// DCRNN with H0=0  =>  R gate dead, H = (1-Z)*tanh(pre_h), out = mean(relu(H)@W_lin)+b_lin.
// Pipeline: deg atomics -> 1/deg -> edge scatter (T1o/T1i accumulators) ->
//           pack weights to WMMA-B layout (f16) -> per-16-node-tile WMMA GEMM
//           (K=96 as 3 x v_wmma_f32_16x16x32_f16 steps: x | T1o | T1i) ->
//           sigmoid/tanh/relu epilogue + global mean reduction.

typedef __attribute__((ext_vector_type(16))) _Float16 v16h;
typedef __attribute__((ext_vector_type(8)))  float    v8f;

#define F 32          // F_IN == F_OUT == 32
#define WAVES_PER_BLOCK 8

// ---------------------------------------------------------------- zero fill
__global__ void k_zero(float* __restrict__ p, long n) {
    long i = (long)blockIdx.x * blockDim.x + threadIdx.x;
    long stride = (long)gridDim.x * blockDim.x;
    for (; i < n; i += stride) p[i] = 0.0f;
}

// ------------------------------------------------------------- degree sums
__global__ void k_deg(const int* __restrict__ src, const int* __restrict__ dst,
                      const float* __restrict__ w,
                      float* __restrict__ dego, float* __restrict__ degi, int E) {
    int e = blockIdx.x * blockDim.x + threadIdx.x;
    if (e >= E) return;
    float wv = w[e];
    atomicAdd(dego + src[e], wv);
    atomicAdd(degi + dst[e], wv);
}

__global__ void k_inv(float* __restrict__ dego, float* __restrict__ degi, int N) {
    int n = blockIdx.x * blockDim.x + threadIdx.x;
    if (n >= N) return;
    dego[n] = 1.0f / dego[n];
    degi[n] = 1.0f / degi[n];
}

// ------------------------------------------------- edge feature scatter (SpMM)
// One wave per edge, one lane per feature: coalesced 128B row reads of x,
// coalesced 32-float atomic bursts into the accumulators (all L2-resident).
__global__ void k_scatter(const int* __restrict__ src, const int* __restrict__ dst,
                          const float* __restrict__ w, const float* __restrict__ x,
                          float* __restrict__ acco, float* __restrict__ acci, int E) {
    int e = blockIdx.x * WAVES_PER_BLOCK + (threadIdx.x >> 5);
    if (e >= E) return;
    int f = threadIdx.x & 31;
    int s = src[e], d = dst[e];
    float wv = w[e];
    atomicAdd(acco + (long)s * F + f, wv * x[(long)d * F + f]);   // T1o[src] += w*x[dst]
    atomicAdd(acci + (long)d * F + f, wv * x[(long)s * F + f]);   // T1i[dst] += w*x[src]
}

// ---------------------------------------------- pack weights into WMMA-B layout
// Bpk[(g*2+t)*3+s][lane][j]  (f16), g=gate(z/h), t=column half, s=K step.
// B 32x16 layout: lane L, base0=8*(L>=16): elems 0..7 = K base0..base0+7, elems 8..15 = K base0+16..base0+23; N=lane%16.
// Cat rows: K<32 -> W[0,0]+W[1,0] (x term), 32..63 -> W[0,1] (T1o), 64..95 -> W[1,1] (T1i);
// only c<32 of the C=64 axis is live (H part of concat is zero).
__global__ void k_prep(const float* __restrict__ Wz, const float* __restrict__ Wh,
                       _Float16* __restrict__ Bpk) {
    int tid = blockIdx.x * blockDim.x + threadIdx.x;
    if (tid >= 12 * 32 * 16) return;
    int j    = tid & 15;
    int lane = (tid >> 4) & 31;
    int rest = tid >> 9;              // (g*2+t)*3 + s  in 0..11
    int s    = rest % 3;
    int gt   = rest / 3;
    int t    = gt & 1;
    int g    = gt >> 1;
    int c    = (j & 7) + ((j >> 3) << 4) + ((lane >> 4) << 3);   // local K within the 32-wide step
    int n    = t * 16 + (lane & 15);       // output column
    const float* W = g ? Wh : Wz;          // W[d,k,c,f] flat: ((d*2+k)*64 + c)*32 + f
    float v;
    if (s == 0)      v = W[(0 * 64 + c) * F + n] + W[(2 * 64 + c) * F + n];
    else if (s == 1) v = W[(1 * 64 + c) * F + n];
    else             v = W[(3 * 64 + c) * F + n];
    Bpk[(size_t)rest * 512 + lane * 16 + j] = (_Float16)v;
}

// -------------------------------------------- WMMA GEMM + activations + reduce
// One wave per 16-node tile. A 16x32 f16 layout: lane L -> row M=L%16,
// elems 0..7 = K base0..base0+7, elems 8..15 = K base0+16..base0+23, base0 = 8*(L>=16).
__global__ void __launch_bounds__(256)
k_gemm(const float* __restrict__ x,
       const float* __restrict__ acco, const float* __restrict__ acci,
       const float* __restrict__ dego_inv, const float* __restrict__ degi_inv,
       const _Float16* __restrict__ Bpk,
       const float* __restrict__ bz, const float* __restrict__ bh,
       const float* __restrict__ wlin, float* __restrict__ res, int ntiles) {
    int wave = blockIdx.x * WAVES_PER_BLOCK + (threadIdx.x >> 5);
    if (wave >= ntiles) return;                 // wave-uniform: EXEC stays all-ones
    int lane  = threadIdx.x & 31;
    int m     = lane & 15;
    int base0 = (lane >> 4) * 8;
    long node = (long)wave * 16 + m;

    const float* srcs[3]  = { x, acco, acci };
    float        scal[3]  = { 1.0f, dego_inv[node], degi_inv[node] };

    v8f cz0 = {}, cz1 = {}, ch0 = {}, ch1 = {};
#pragma unroll
    for (int s = 0; s < 3; ++s) {
        const float* row = srcs[s] + node * F;
        float sc = scal[s];
        v16h a;
#pragma unroll
        for (int jj = 0; jj < 8; ++jj) a[jj]     = (_Float16)(row[base0 + jj] * sc);
#pragma unroll
        for (int jj = 0; jj < 8; ++jj) a[8 + jj] = (_Float16)(row[base0 + 16 + jj] * sc);

        v16h b0 = *(const v16h*)(Bpk + (size_t)(0 * 3 + s) * 512 + lane * 16);  // z, cols 0-15
        v16h b1 = *(const v16h*)(Bpk + (size_t)(1 * 3 + s) * 512 + lane * 16);  // z, cols 16-31
        v16h b2 = *(const v16h*)(Bpk + (size_t)(2 * 3 + s) * 512 + lane * 16);  // h, cols 0-15
        v16h b3 = *(const v16h*)(Bpk + (size_t)(3 * 3 + s) * 512 + lane * 16);  // h, cols 16-31
        cz0 = __builtin_amdgcn_wmma_f32_16x16x32_f16(false, a, false, b0, (short)0, cz0, false, false);
        cz1 = __builtin_amdgcn_wmma_f32_16x16x32_f16(false, a, false, b1, (short)0, cz1, false, false);
        ch0 = __builtin_amdgcn_wmma_f32_16x16x32_f16(false, a, false, b2, (short)0, ch0, false, false);
        ch1 = __builtin_amdgcn_wmma_f32_16x16x32_f16(false, a, false, b3, (short)0, ch1, false, false);
    }

    // C/D layout: lane holds column N = lane%16 for 8 node-rows; result is a global
    // mean, so only the column index matters for bias / W_lin.
    int   col = lane & 15;
    float bz0 = bz[col], bz1 = bz[16 + col];
    float bh0 = bh[col], bh1 = bh[16 + col];
    float wl0 = wlin[col], wl1 = wlin[16 + col];
    float lsum = 0.0f;
#pragma unroll
    for (int jj = 0; jj < 8; ++jj) {
        float z  = 1.0f / (1.0f + expf(-(cz0[jj] + bz0)));
        float ht = tanhf(ch0[jj] + bh0);
        lsum += fmaxf((1.0f - z) * ht, 0.0f) * wl0;
        z  = 1.0f / (1.0f + expf(-(cz1[jj] + bz1)));
        ht = tanhf(ch1[jj] + bh1);
        lsum += fmaxf((1.0f - z) * ht, 0.0f) * wl1;
    }
#pragma unroll
    for (int d = 16; d >= 1; d >>= 1) lsum += __shfl_xor(lsum, d, 32);
    if (lane == 0) atomicAdd(res, lsum);
}

__global__ void k_final(const float* __restrict__ res, const float* __restrict__ blin,
                        float* __restrict__ out, float invN) {
    if (blockIdx.x == 0 && threadIdx.x == 0) out[0] = res[0] * invN + blin[0];
}

// ---------------------------------------------------------------------------
extern "C" void kernel_launch(void* const* d_in, const int* in_sizes, int n_in,
                              void* d_out, int out_size, void* d_ws, size_t ws_size,
                              hipStream_t stream) {
    const float* x    = (const float*)d_in[0];
    const float* ew   = (const float*)d_in[1];
    const float* Wz   = (const float*)d_in[2];
    const float* bz   = (const float*)d_in[3];
    // d_in[4]=W_r, d_in[5]=b_r: dead (H0 == 0 -> R only feeds H*R == 0)
    const float* Wh   = (const float*)d_in[6];
    const float* bh   = (const float*)d_in[7];
    const float* wlin = (const float*)d_in[8];
    const float* blin = (const float*)d_in[9];
    const int*   eidx = (const int*)d_in[10];

    const int N = in_sizes[0] / F;
    const int E = in_sizes[1];
    const int* src = eidx;
    const int* dst = eidx + E;

    // workspace carve-up (256B aligned)
    char* ws = (char*)d_ws;
    size_t off = 0;
    auto carve = [&](size_t bytes) -> char* {
        char* p = ws + off;
        off = (off + bytes + 255) & ~(size_t)255;
        return p;
    };
    float*     deg_o = (float*)carve((size_t)N * 4);
    float*     deg_i = (float*)carve((size_t)N * 4);
    float*     acc_o = (float*)carve((size_t)N * F * 4);
    float*     acc_i = (float*)carve((size_t)N * F * 4);
    float*     res   = (float*)carve(256);
    size_t zero_bytes = off;                    // deg_o .. res must start at 0
    _Float16*  bpk   = (_Float16*)carve(12 * 512 * sizeof(_Float16));
    (void)ws_size;

    long zn = (long)(zero_bytes / 4);
    k_zero<<<2048, 256, 0, stream>>>((float*)ws, zn);

    k_deg<<<(E + 255) / 256, 256, 0, stream>>>(src, dst, ew, deg_o, deg_i, E);
    k_inv<<<(N + 255) / 256, 256, 0, stream>>>(deg_o, deg_i, N);
    k_scatter<<<(E + WAVES_PER_BLOCK - 1) / WAVES_PER_BLOCK, 256, 0, stream>>>(
        src, dst, ew, x, acc_o, acc_i, E);
    k_prep<<<(12 * 512 + 255) / 256, 256, 0, stream>>>(Wz, Wh, bpk);

    int ntiles = N / 16;   // N = 100000 is a multiple of 16: no partial WMMA tiles
    k_gemm<<<(ntiles + WAVES_PER_BLOCK - 1) / WAVES_PER_BLOCK, 256, 0, stream>>>(
        x, acc_o, acc_i, deg_o, deg_i, bpk, bz, bh, wlin, res, ntiles);

    k_final<<<1, 64, 0, stream>>>(res, blin, (float*)d_out, 1.0f / (float)N);
}
